// CausalSelfAttention_16389595202213
// MI455X (gfx1250) — compile-verified, matrix operand order checked by reading, not run on hardware
//
#include <hip/hip_runtime.h>

// ---------------------------------------------------------------------------
// CDNA5 (gfx1250) causal self-attention, bf16 WMMA pipeline:
//   1) convert x -> bf16; W_attn, W_proj -> bf16 transposed ([N][K])
//   2) QKV GEMM (x @ W_attn + b), scatter to q/k/v [B,H,T,D] bf16
//   3) flash attention (causal, online softmax), y [B,T,C] bf16
//   4) out = y @ W_proj + b_proj  (f32 output)
// All matmuls use v_wmma_f32_16x16x32_bf16 (f32 accumulate).
// Global->LDS tile fills use GLOBAL_LOAD_ASYNC_TO_LDS_B128 + s_wait_asynccnt
// with double-buffered LDS when the toolchain exposes the builtin.
// ---------------------------------------------------------------------------

typedef __bf16 bf16;
typedef __attribute__((ext_vector_type(16))) __bf16 v16bf;
typedef __attribute__((ext_vector_type(8)))  __bf16 v8bf;
typedef __attribute__((ext_vector_type(8)))  float  v8f;
typedef __attribute__((ext_vector_type(4)))  int    v4i;

#if defined(__AMDGCN__) && __has_builtin(__builtin_amdgcn_global_load_async_to_lds_b128)
#define LDS_ASYNC 1
#else
#define LDS_ASYNC 0
#endif

#if LDS_ASYNC
typedef __attribute__((address_space(3))) v4i las_v4i;
#define WAIT_ASYNC(n) asm volatile("s_wait_asynccnt %0" ::"i"(n) : "memory")
#else
#define WAIT_ASYNC(n)
#endif

// Copy 16 bytes global -> LDS (async path bypasses VGPRs, ASYNCcnt-tracked).
__device__ __forceinline__ void copy16_g2l(const bf16* src, bf16* dst_lds) {
#if LDS_ASYNC
  __builtin_amdgcn_global_load_async_to_lds_b128((v4i*)src, (las_v4i*)dst_lds, 0, 0);
#else
  *(v8bf*)dst_lds = *(const v8bf*)src;
#endif
}

__device__ __forceinline__ v8f vzero8f() {
  v8f z;
#pragma unroll
  for (int i = 0; i < 8; ++i) z[i] = 0.0f;
  return z;
}

// Pack one 16x32 WMMA A/B fragment for this lane from a contiguous 32-element
// K-run at row32. Per ISA 7.12.2: elems 0..7 = K[half*8 .. +7],
// elems 8..15 = K[16+half*8 .. +7].
__device__ __forceinline__ v16bf load_frag32(const bf16* row32, int hf) {
  v8bf lo = *(const v8bf*)(row32 + hf * 8);
  v8bf hi = *(const v8bf*)(row32 + 16 + hf * 8);
  v16bf r;
#pragma unroll
  for (int i = 0; i < 8; ++i) { r[i] = lo[i]; r[8 + i] = hi[i]; }
  return r;
}

__device__ __forceinline__ v8f wmma_bf16(v16bf a, v16bf b, v8f c) {
  return __builtin_amdgcn_wmma_f32_16x16x32_bf16(false, a, false, b,
                                                 (short)0, c, false, false);
}

// ---------------------------------------------------------------------------
// Elementwise converts
// ---------------------------------------------------------------------------
__global__ void f32_to_bf16_kernel(const float* __restrict__ in,
                                   bf16* __restrict__ out, int n) {
  int i = blockIdx.x * blockDim.x + threadIdx.x;
  if (i < n) out[i] = (bf16)in[i];
}

// in: [K][N] f32 (row-major), out: [N][K] bf16
__global__ void transpose_to_bf16_kernel(const float* __restrict__ in,
                                         bf16* __restrict__ out,
                                         int K, int N) {
  int i = blockIdx.x * blockDim.x + threadIdx.x;
  if (i < N * K) {
    int n = i / K, kk = i - n * K;
    out[i] = (bf16)in[(size_t)kk * N + n];
  }
}

// ---------------------------------------------------------------------------
// QKV GEMM: A [8192][1024] bf16 (x), BT [3072][1024] bf16 (W_attn^T)
// scatter to q/k/v [B=4,H=16,T=2048,D=64] bf16
// Double-buffered LDS, async global->LDS fills.
// ---------------------------------------------------------------------------
#define GK 1024

__global__ __launch_bounds__(256) void qkv_gemm_kernel(
    const bf16* __restrict__ A, const bf16* __restrict__ BT,
    const float* __restrict__ bias,
    bf16* __restrict__ qd, bf16* __restrict__ kd, bf16* __restrict__ vd) {
  __shared__ bf16 sA[2][128][40];  // 32 K + 8 pad (16B) -> conflict-free b128
  __shared__ bf16 sB[2][128][40];
  const int tid = threadIdx.x;
  const int n0 = blockIdx.x * 128;
  const int m0 = blockIdx.y * 128;
  const int w = tid >> 5, lane = tid & 31, hf = lane >> 4, nl = lane & 15;
  const int waveM = w >> 1, waveN = w & 1;

  auto load_tile = [&](int kt, int buf) {
#pragma unroll
    for (int c = 0; c < 2; ++c) {
      int ch = tid + 256 * c;                 // 512 chunks of 8 bf16 per matrix
      int row = ch >> 2, k8 = (ch & 3) * 8;
      copy16_g2l(A + (size_t)(m0 + row) * GK + kt + k8, &sA[buf][row][k8]);
      copy16_g2l(BT + (size_t)(n0 + row) * GK + kt + k8, &sB[buf][row][k8]);
    }
  };

  v8f acc[2][4];
#pragma unroll
  for (int i = 0; i < 2; ++i)
#pragma unroll
    for (int j = 0; j < 4; ++j) acc[i][j] = vzero8f();

  load_tile(0, 0);                 // prologue: tile 0 in flight
  const int NT = GK / 32;
  for (int it = 0; it < NT; ++it) {
    __syncthreads();               // everyone done reading buf[(it+1)&1]
    if (it + 1 < NT) {
      load_tile((it + 1) * 32, (it + 1) & 1);
      WAIT_ASYNC(4);               // 4/thread in flight for next tile; cur done
    } else {
      WAIT_ASYNC(0);
    }
    __syncthreads();               // all waves' tile-it data landed

    const int bs = it & 1;
    v16bf bfr[4];
#pragma unroll
    for (int nf = 0; nf < 4; ++nf)
      bfr[nf] = load_frag32(&sB[bs][waveN * 64 + nf * 16 + nl][0], hf);
#pragma unroll
    for (int mf = 0; mf < 2; ++mf) {
      v16bf a = load_frag32(&sA[bs][waveM * 32 + mf * 16 + nl][0], hf);
#pragma unroll
      for (int nf = 0; nf < 4; ++nf)
        acc[mf][nf] = wmma_bf16(a, bfr[nf], acc[mf][nf]);
    }
  }

#pragma unroll
  for (int mf = 0; mf < 2; ++mf)
#pragma unroll
    for (int nf = 0; nf < 4; ++nf) {
      int ncol = n0 + waveN * 64 + nf * 16 + nl;   // 0..3071
      float bv = bias[ncol];
      int which = ncol >> 10, cc = ncol & 1023;
      int h = cc >> 6, d = cc & 63;
      bf16* dst = which == 0 ? qd : (which == 1 ? kd : vd);
#pragma unroll
      for (int e = 0; e < 8; ++e) {
        int mrow = m0 + waveM * 32 + mf * 16 + e + 8 * hf;  // b*T + t
        int b = mrow >> 11, t = mrow & 2047;
        dst[(((size_t)b * 16 + h) * 2048 + t) * 64 + d] = (bf16)(acc[mf][nf][e] + bv);
      }
    }
}

// ---------------------------------------------------------------------------
// Flash attention (causal). grid = (B*H=64, T/128=16), 256 threads = 8 waves.
// Wave w handles query rows q0 = qt*128 + w*16 .. +15. KV blocks of 64,
// double-buffered; K tile async, V tile transposed through VGPRs.
// ---------------------------------------------------------------------------
__global__ __launch_bounds__(256) void flash_attn_kernel(
    const bf16* __restrict__ q, const bf16* __restrict__ k,
    const bf16* __restrict__ v, bf16* __restrict__ y) {
  __shared__ bf16 sK[2][64][72];    // [key][d]   (K^T B-frags contiguous)
  __shared__ bf16 sVT[2][64][72];   // [d][key]   (V B-frags contiguous)
  __shared__ bf16 sP[8][16][72];    // per-wave P staging (C-layout -> A-layout)
  const int bh = blockIdx.x;        // b*16 + h
  const int qt = blockIdx.y;
  const int tid = threadIdx.x;
  const int w = tid >> 5, lane = tid & 31, hf = lane >> 4, nl = lane & 15;
  const int q0 = qt * 128 + w * 16;
  const size_t base = (size_t)bh * 2048 * 64;

  auto load_kv = [&](int kb, int buf) {
#pragma unroll
    for (int c = 0; c < 2; ++c) {
      int ch = tid + 256 * c;               // 512 chunks of 8
      int row = ch >> 3, c8 = (ch & 7) * 8;
      copy16_g2l(k + base + (size_t)(kb + row) * 64 + c8, &sK[buf][row][c8]);
      v8bf dv = *(const v8bf*)(v + base + (size_t)(kb + row) * 64 + c8);
#pragma unroll
      for (int i = 0; i < 8; ++i) sVT[buf][c8 + i][row] = dv[i];
    }
  };

  // Q fragments (16 rows x 64 d) live in registers for the whole kernel.
  v16bf qf[2];
  {
    const bf16* row = q + base + (size_t)(q0 + nl) * 64;
#pragma unroll
    for (int f = 0; f < 2; ++f) qf[f] = load_frag32(row + f * 32, hf);
  }

  float mrow[8], lrow[8];
  v8f acc[4];
#pragma unroll
  for (int e = 0; e < 8; ++e) { mrow[e] = -1e30f; lrow[e] = 0.0f; }
#pragma unroll
  for (int nf = 0; nf < 4; ++nf) acc[nf] = vzero8f();

  const int njb = 2 * (qt + 1);
  load_kv(0, 0);                    // prologue
  for (int jb = 0; jb < njb; ++jb) {
    const int kb = jb * 64;
    __syncthreads();                // prev compute done; safe to refill
    if (jb + 1 < njb) {
      load_kv((jb + 1) * 64, (jb + 1) & 1);
      WAIT_ASYNC(2);                // 2/thread in flight for next K tile
    } else {
      WAIT_ASYNC(0);
    }
    __syncthreads();                // tile jb fully resident

    const int bs = jb & 1;
    if (kb <= q0 + 15) {            // wave-uniform: EXEC stays all-ones
      // S = (q @ k^T) : 16 x 64, f32
      v8f S[4];
#pragma unroll
      for (int nf = 0; nf < 4; ++nf) S[nf] = vzero8f();
#pragma unroll
      for (int kf = 0; kf < 2; ++kf)
#pragma unroll
        for (int nf = 0; nf < 4; ++nf) {
          v16bf bb = load_frag32(&sK[bs][nf * 16 + nl][kf * 32], hf);
          S[nf] = wmma_bf16(qf[kf], bb, S[nf]);
        }

      // scale + causal mask + online softmax (rows live across 16 lanes)
      const float scale = 0.125f;   // 1/sqrt(64)
#pragma unroll
      for (int e = 0; e < 8; ++e) {
        const int qrow = q0 + e + 8 * hf;
        float bmax = -1e30f;
#pragma unroll
        for (int nf = 0; nf < 4; ++nf) {
          int key = kb + nf * 16 + nl;
          float s = S[nf][e] * scale;
          if (key > qrow) s = -1e30f;
          S[nf][e] = s;
          bmax = fmaxf(bmax, s);
        }
#pragma unroll
        for (int off = 8; off >= 1; off >>= 1)
          bmax = fmaxf(bmax, __shfl_xor(bmax, off, 16));
        const float mnew = fmaxf(mrow[e], bmax);
        const float alpha = __expf(mrow[e] - mnew);
        mrow[e] = mnew;
        float psum = 0.0f;
#pragma unroll
        for (int nf = 0; nf < 4; ++nf) {
          float p = __expf(S[nf][e] - mnew);
          S[nf][e] = p;
          psum += p;
        }
#pragma unroll
        for (int off = 8; off >= 1; off >>= 1)
          psum += __shfl_xor(psum, off, 16);
        lrow[e] = lrow[e] * alpha + psum;
#pragma unroll
        for (int nf = 0; nf < 4; ++nf) acc[nf][e] *= alpha;
      }

      // P: C-layout -> LDS -> A-layout (same wave, so just a DS wait)
#pragma unroll
      for (int nf = 0; nf < 4; ++nf)
#pragma unroll
        for (int e = 0; e < 8; ++e)
          sP[w][e + 8 * hf][nf * 16 + nl] = (bf16)S[nf][e];
      asm volatile("s_wait_dscnt 0" ::: "memory");

      // acc += P @ V
#pragma unroll
      for (int kf = 0; kf < 2; ++kf) {
        v16bf a = load_frag32(&sP[w][nl][kf * 32], hf);
#pragma unroll
        for (int nf = 0; nf < 4; ++nf) {
          v16bf bb = load_frag32(&sVT[bs][nf * 16 + nl][kf * 32], hf);
          acc[nf] = wmma_bf16(a, bb, acc[nf]);
        }
      }
    }
  }

  // normalize + store y [B,T,C] bf16, C index = h*64 + d
  const int b = bh >> 4, h = bh & 15;
#pragma unroll
  for (int e = 0; e < 8; ++e) {
    const float inv = 1.0f / lrow[e];
    const int t = q0 + e + 8 * hf;
#pragma unroll
    for (int nf = 0; nf < 4; ++nf) {
      int d = nf * 16 + nl;
      y[((size_t)b * 2048 + t) * 1024 + h * 64 + d] = (bf16)(acc[nf][e] * inv);
    }
  }
}

// ---------------------------------------------------------------------------
// Output projection: A = y bf16 [8192][1024], BT = W_proj^T bf16 [1024][1024]
// out f32 [8192][1024]. Same double-buffered async structure.
// ---------------------------------------------------------------------------
__global__ __launch_bounds__(256) void proj_gemm_kernel(
    const bf16* __restrict__ A, const bf16* __restrict__ BT,
    const float* __restrict__ bias, float* __restrict__ out) {
  __shared__ bf16 sA[2][128][40];
  __shared__ bf16 sB[2][128][40];
  const int tid = threadIdx.x;
  const int n0 = blockIdx.x * 128;
  const int m0 = blockIdx.y * 128;
  const int w = tid >> 5, lane = tid & 31, hf = lane >> 4, nl = lane & 15;
  const int waveM = w >> 1, waveN = w & 1;

  auto load_tile = [&](int kt, int buf) {
#pragma unroll
    for (int c = 0; c < 2; ++c) {
      int ch = tid + 256 * c;
      int row = ch >> 2, k8 = (ch & 3) * 8;
      copy16_g2l(A + (size_t)(m0 + row) * GK + kt + k8, &sA[buf][row][k8]);
      copy16_g2l(BT + (size_t)(n0 + row) * GK + kt + k8, &sB[buf][row][k8]);
    }
  };

  v8f acc[2][4];
#pragma unroll
  for (int i = 0; i < 2; ++i)
#pragma unroll
    for (int j = 0; j < 4; ++j) acc[i][j] = vzero8f();

  load_tile(0, 0);
  const int NT = GK / 32;
  for (int it = 0; it < NT; ++it) {
    __syncthreads();
    if (it + 1 < NT) {
      load_tile((it + 1) * 32, (it + 1) & 1);
      WAIT_ASYNC(4);
    } else {
      WAIT_ASYNC(0);
    }
    __syncthreads();

    const int bs = it & 1;
    v16bf bfr[4];
#pragma unroll
    for (int nf = 0; nf < 4; ++nf)
      bfr[nf] = load_frag32(&sB[bs][waveN * 64 + nf * 16 + nl][0], hf);
#pragma unroll
    for (int mf = 0; mf < 2; ++mf) {
      v16bf a = load_frag32(&sA[bs][waveM * 32 + mf * 16 + nl][0], hf);
#pragma unroll
      for (int nf = 0; nf < 4; ++nf)
        acc[mf][nf] = wmma_bf16(a, bfr[nf], acc[mf][nf]);
    }
  }

#pragma unroll
  for (int mf = 0; mf < 2; ++mf)
#pragma unroll
    for (int nf = 0; nf < 4; ++nf) {
      int ncol = n0 + waveN * 64 + nf * 16 + nl;
      float bv = bias[ncol];
#pragma unroll
      for (int e = 0; e < 8; ++e) {
        int mrow = m0 + waveM * 32 + mf * 16 + e + 8 * hf;
        out[(size_t)mrow * 1024 + ncol] = acc[mf][nf][e] + bv;
      }
    }
}

// ---------------------------------------------------------------------------
// Launcher
// ---------------------------------------------------------------------------
extern "C" void kernel_launch(void* const* d_in, const int* in_sizes, int n_in,
                              void* d_out, int out_size, void* d_ws, size_t ws_size,
                              hipStream_t stream) {
  (void)in_sizes; (void)n_in; (void)out_size; (void)ws_size;
  const float* x  = (const float*)d_in[0];   // [4,2048,1024]
  const float* Wa = (const float*)d_in[1];   // [1024,3072]
  const float* ba = (const float*)d_in[2];   // [3072]
  const float* Wp = (const float*)d_in[3];   // [1024,1024]
  const float* bp = (const float*)d_in[4];   // [1024]
  float* out = (float*)d_out;                // [4,2048,1024] f32

  const size_t SZ_X  = (size_t)8192 * 1024;  // 8,388,608
  const size_t SZ_WA = (size_t)3072 * 1024;
  const size_t SZ_WP = (size_t)1024 * 1024;

  bf16* xb  = (bf16*)d_ws;            // x bf16
  bf16* wat = xb  + SZ_X;             // W_attn^T bf16 [3072][1024]
  bf16* wpt = wat + SZ_WA;            // W_proj^T bf16 [1024][1024]
  bf16* qd  = wpt + SZ_WP;            // q [B,H,T,D]
  bf16* kd  = qd  + SZ_X;             // k
  bf16* vd  = kd  + SZ_X;             // v
  bf16* yb  = vd  + SZ_X;             // attention output bf16 [B,T,C]

  f32_to_bf16_kernel<<<(int)((SZ_X + 255) / 256), 256, 0, stream>>>(x, xb, (int)SZ_X);
  transpose_to_bf16_kernel<<<(int)((SZ_WA + 255) / 256), 256, 0, stream>>>(Wa, wat, 1024, 3072);
  transpose_to_bf16_kernel<<<(int)((SZ_WP + 255) / 256), 256, 0, stream>>>(Wp, wpt, 1024, 1024);

  qkv_gemm_kernel<<<dim3(24, 64), 256, 0, stream>>>(xb, wat, ba, qd, kd, vd);
  flash_attn_kernel<<<dim3(64, 16), 256, 0, stream>>>(qd, kd, vd, yb);
  proj_gemm_kernel<<<dim3(8, 64), 256, 0, stream>>>(yb, wpt, bp, out);
}
